// BiasFreeClifordNetBlock_51573967290676
// MI455X (gfx1250) — matrix operand order repaired and checked by reading, not compiled, not run on hardware
//
#include <hip/hip_runtime.h>
#include <hip/hip_bf16.h>
#include <math.h>

// ---------------------------------------------------------------------------
// BiasFreeCliffordNetBlock for MI455X (gfx1250), bf16 WMMA pipeline.
// B=4, H=128, W=128, C=256.
// Round 2: LDS-staged weight tiles via GLOBAL_LOAD_ASYNC_TO_LDS (ASYNCcnt),
// WMMA fed from ds_load_b128 for B and streamed global A.
// ---------------------------------------------------------------------------

typedef __bf16 bf16;
typedef __attribute__((ext_vector_type(16))) __bf16 v16bf;
typedef __attribute__((ext_vector_type(8)))  __bf16 v8bf;
typedef __attribute__((ext_vector_type(8)))  float  v8f;

#define NB   4
#define HH   128
#define WWID 128
#define CC   256
#define NPIX (NB * HH * WWID)   // 65536
#define PPB  (HH * WWID)        // 16384 pixels per batch

// ---------------- helpers ----------------

static __device__ __forceinline__ bf16 f2bf(float f) {
  unsigned u = __builtin_bit_cast(unsigned, f);
  unsigned r = u + 0x7FFFu + ((u >> 16) & 1u);   // round-to-nearest-even
  unsigned short h = (unsigned short)(r >> 16);
  return __builtin_bit_cast(__bf16, h);
}

static __device__ __forceinline__ float bf2f(bf16 b) {
  unsigned short h = __builtin_bit_cast(unsigned short, b);
  return __builtin_bit_cast(float, (unsigned)h << 16);
}

static __device__ __forceinline__ v16bf join8(v8bf a, v8bf b) {
  return __builtin_shufflevector(a, b, 0,1,2,3,4,5,6,7,8,9,10,11,12,13,14,15);
}

static __device__ __forceinline__ v8f vzero8() {
  v8f z = {0.f,0.f,0.f,0.f,0.f,0.f,0.f,0.f};
  return z;
}

// LDS byte offset of a __shared__ object: low 32 bits of its generic address
// (shared aperture base has zero low bits; flat->LDS mapping truncates).
static __device__ __forceinline__ unsigned lds_off(const void* p) {
  return (unsigned)(unsigned long long)p;
}

// Async global->LDS 16B copy (per-lane). Tracked by ASYNCcnt.
static __device__ __forceinline__ void async_copy16(unsigned lds_byte,
                                                    const void* gptr) {
  unsigned long long ga = (unsigned long long)gptr;
  asm volatile("global_load_async_to_lds_b128 %0, %1, off"
               :: "v"(lds_byte), "v"(ga) : "memory");
}

static __device__ __forceinline__ void async_wait0() {
  asm volatile("s_wait_asynccnt 0" ::: "memory");
}

// A fragment (16x32 bf16, MxK) per ISA 7.12.2:
// lane l: m = l%16, kh = l/16; elems 0..7 = K[8kh .. 8kh+7], elems 8..15 = K[16+8kh ..].
static __device__ __forceinline__ v16bf load_a_frag(const bf16* __restrict__ A,
                                                    int lda, int row0, int k0, int lane) {
  int m  = row0 + (lane & 15);
  int kh = lane >> 4;
  const bf16* p = A + (size_t)m * lda + k0 + 8 * kh;
  return join8(*(const v8bf*)p, *(const v8bf*)(p + 16));
}

// B fragment (32x16 bf16, KxN) from transposed weights BT[N][K]:
// lane l: n = l%16, kh = l/16; elems 0..7 = BT[n][k0+8kh ..], elems 8..15 = BT[n][k0+16+8kh ..].
static __device__ __forceinline__ v16bf load_b_frag(const bf16* __restrict__ BT,
                                                    int ldb, int n0, int k0, int lane) {
  int n  = n0 + (lane & 15);
  int kh = lane >> 4;
  const bf16* p = BT + (size_t)n * ldb + k0 + 8 * kh;
  return join8(*(const v8bf*)p, *(const v8bf*)(p + 16));
}

// ---------------- kernel 0: transpose weights to bf16 [N][K] ----------------

__global__ void __launch_bounds__(256)
k_prep(const float* __restrict__ w_det, const float* __restrict__ w_gl,
       const float* __restrict__ w_gg,  const float* __restrict__ w_gate,
       bf16* __restrict__ wdetT, bf16* __restrict__ wlT,
       bf16* __restrict__ wgT,   bf16* __restrict__ wgtT) {
  int stride = gridDim.x * blockDim.x;
  int tid = blockIdx.x * blockDim.x + threadIdx.x;
  for (int i = tid; i < CC * CC; i += stride) {       // w_det: 256x256
    int n = i >> 8, k = i & 255;
    wdetT[i] = f2bf(w_det[(size_t)k * CC + n]);
  }
  for (int i = tid; i < CC * 2048; i += stride) {     // w_geo_local: 2048x256
    int n = i >> 11, k = i & 2047;
    wlT[i] = f2bf(w_gl[(size_t)k * CC + n]);
  }
  for (int i = tid; i < CC * 1024; i += stride) {     // w_geo_global: 1024x256
    int n = i >> 10, k = i & 1023;
    wgT[i] = f2bf(w_gg[(size_t)k * CC + n]);
  }
  for (int i = tid; i < CC * 512; i += stride) {      // w_gate: 512x256
    int n = i >> 9, k = i & 511;
    wgtT[i] = f2bf(w_gate[(size_t)k * CC + n]);
  }
}

// ---------------- kernel 1: LayerNorm (scale-only) -> xn bf16 ----------------

__global__ void __launch_bounds__(256)
k_ln(const float* __restrict__ x, const float* __restrict__ gamma,
     bf16* __restrict__ xn) {
  int wave = threadIdx.x >> 5, lane = threadIdx.x & 31;
  int pix  = blockIdx.x * 8 + wave;                 // one wave per pixel
  const float* xp = x + (size_t)pix * CC;
  float v[8];
  float s = 0.f;
#pragma unroll
  for (int j = 0; j < 8; ++j) { v[j] = xp[lane + 32 * j]; s += v[j]; }
#pragma unroll
  for (int o = 16; o > 0; o >>= 1) s += __shfl_xor(s, o, 32);
  float mu = s * (1.f / 256.f);
  float q = 0.f;
#pragma unroll
  for (int j = 0; j < 8; ++j) { float d = v[j] - mu; q += d * d; }
#pragma unroll
  for (int o = 16; o > 0; o >>= 1) q += __shfl_xor(q, o, 32);
  float inv = rsqrtf(q * (1.f / 256.f) + 1e-6f);
  bf16* op = xn + (size_t)pix * CC;
#pragma unroll
  for (int j = 0; j < 8; ++j) {
    int c = lane + 32 * j;
    op[c] = f2bf((v[j] - mu) * inv * gamma[c]);
  }
}

// ---------------- kernel 2: deterministic per-batch spatial mean of xn ------

__global__ void __launch_bounds__(256)
k_mean(const bf16* __restrict__ xn, float* __restrict__ means) {
  // grid = NB*16; block handles (batch, 16 channels)
  int b  = blockIdx.x >> 4;
  int c0 = (blockIdx.x & 15) * 16;
  int c  = c0 + (threadIdx.x & 15);
  int pg = threadIdx.x >> 4;                        // 0..15
  const bf16* base = xn + (size_t)b * PPB * CC;
  float s = 0.f;
  for (int p = pg; p < PPB; p += 16) s += bf2f(base[(size_t)p * CC + c]);
  __shared__ float red[256];
  red[threadIdx.x] = s;
  __syncthreads();
  if (threadIdx.x < 16) {
    float t = 0.f;
#pragma unroll
    for (int g = 0; g < 16; ++g) t += red[g * 16 + threadIdx.x];
    means[b * CC + c0 + threadIdx.x] = t * (1.f / (float)PPB);
  }
}

// ---------------- kernel 3: z_det = xn @ w_det (WMMA bf16) -------------------
// B tile (64 cols x 256 K, 32KB) staged to LDS once via async-to-LDS.

__global__ void __launch_bounds__(256)
k_gemm_det(const bf16* __restrict__ xn, const bf16* __restrict__ wdetT,
           bf16* __restrict__ zdet) {
  __shared__ bf16 sW[64 * CC];                      // 32KB: wdetT rows nc0..nc0+63
  int tid  = threadIdx.x;
  int wave = tid >> 5, lane = tid & 31;
  int row0 = blockIdx.x * 128 + wave * 16;
  int nc0  = blockIdx.y * 64;
  {
    const bf16* src = wdetT + (size_t)nc0 * CC;     // contiguous 64*256 bf16
    unsigned l0 = lds_off(&sW[0]);
#pragma unroll
    for (int i = tid; i < 64 * CC / 8; i += 256)
      async_copy16(l0 + i * 16, src + i * 8);
    async_wait0();
  }
  __syncthreads();

  v8f acc[4] = {vzero8(), vzero8(), vzero8(), vzero8()};
  for (int k0 = 0; k0 < CC; k0 += 32) {
    if (k0 + 32 < CC)
      __builtin_prefetch(xn + (size_t)(row0 + (lane & 15)) * CC + k0 + 32, 0, 1);
    v16bf a = load_a_frag(xn, CC, row0, k0, lane);
#pragma unroll
    for (int t = 0; t < 4; ++t) {
      v16bf b = load_b_frag(sW, CC, 16 * t, k0, lane);   // from LDS
      acc[t] = __builtin_amdgcn_wmma_f32_16x16x32_bf16(false, a, false, b,
                                                       (short)0, acc[t], false, false);
    }
  }
  int khl = lane >> 4, nl = lane & 15;
#pragma unroll
  for (int t = 0; t < 4; ++t)
#pragma unroll
    for (int v = 0; v < 8; ++v) {
      int m = row0 + v + 8 * khl;
      int n = nc0 + 16 * t + nl;
      zdet[(size_t)m * CC + n] = f2bf(acc[t][v]);
    }
}

// ---------------- depthwise 3x3 (SAME, zero pad), 4 channels/thread ----------

static __device__ __forceinline__ void dw3x3(const bf16* __restrict__ in,
                                             const float* __restrict__ kern,
                                             int pix, int c0, float acc[4]) {
  int wx = pix & (WWID - 1);
  int hy = (pix >> 7) & (HH - 1);
  int b  = pix >> 14;
  acc[0] = acc[1] = acc[2] = acc[3] = 0.f;
#pragma unroll
  for (int dy = -1; dy <= 1; ++dy) {
    int hh = hy + dy;
    if (hh < 0 || hh >= HH) continue;
#pragma unroll
    for (int dx = -1; dx <= 1; ++dx) {
      int wn = wx + dx;
      if (wn < 0 || wn >= WWID) continue;
      const bf16* p  = in + (((size_t)((b * HH + hh) * WWID + wn)) * CC + c0);
      const float* kk = kern + ((dy + 1) * 3 + (dx + 1)) * CC + c0;
#pragma unroll
      for (int j = 0; j < 4; ++j) acc[j] += bf2f(p[j]) * kk[j];
    }
  }
}

__global__ void __launch_bounds__(256)
k_dw1(const bf16* __restrict__ xn, const float* __restrict__ dw1,
      bf16* __restrict__ t1) {
  int gid = blockIdx.x * 256 + threadIdx.x;
  int c0  = (gid & 63) * 4;
  int pix = gid >> 6;
  float a[4];
  dw3x3(xn, dw1, pix, c0, a);
  bf16* o = t1 + (size_t)pix * CC + c0;
#pragma unroll
  for (int j = 0; j < 4; ++j) o[j] = f2bf(a[j]);
}

__global__ void __launch_bounds__(256)
k_dw2(const bf16* __restrict__ t1, const float* __restrict__ dw2,
      const float* __restrict__ bn_g, const float* __restrict__ bn_m,
      const float* __restrict__ bn_v, const bf16* __restrict__ zdet,
      bf16* __restrict__ zctx) {
  int gid = blockIdx.x * 256 + threadIdx.x;
  int c0  = (gid & 63) * 4;
  int pix = gid >> 6;
  float a[4];
  dw3x3(t1, dw2, pix, c0, a);
  bf16* o = zctx + (size_t)pix * CC + c0;
  const bf16* zd = zdet + (size_t)pix * CC + c0;
#pragma unroll
  for (int j = 0; j < 4; ++j) {
    int c = c0 + j;
    float sc  = bn_g[c] * rsqrtf(bn_v[c] + 1e-3f);
    float val = (a[j] - bn_m[c]) * sc;
    val = val / (1.f + __expf(-val));              // SiLU
    o[j] = f2bf(val - bf2f(zd[j]));                // ctx_mode='diff'
  }
}

// ---------------- kernel 6: geometric-product features + projections ---------
// g_feat = feats_local(zdet, zctx) @ Wl + feats_global(zdet, mean-zdet) @ Wg
// feats column j: si=j>>9 (shift=1<<si), typ=(j>>8)&1, c=j&255, cs=(c-shift)&255
//   sym  : a[c]*b[cs] + a[cs]*b[c]
//   anti : a[c]*b[cs] - a[cs]*b[c]

__global__ void __launch_bounds__(256)
k_geo(const bf16* __restrict__ zdet, const bf16* __restrict__ zctx,
      const float* __restrict__ means, const bf16* __restrict__ wlT,
      const bf16* __restrict__ wgT, bf16* __restrict__ gfeat) {
  __shared__ bf16 sA[32 * CC];   // zdet rows  (16KB)
  __shared__ bf16 sB[32 * CC];   // zctx rows  (16KB)
  __shared__ float sM[CC];       // batch spatial mean of xn
  __shared__ bf16 sF[32 * 32];   // feature tile (32 rows x 32 K)

  int tid  = threadIdx.x;
  int row0 = blockIdx.x * 32;
  {
    const bf16* srcA = zdet + (size_t)row0 * CC;
    const bf16* srcB = zctx + (size_t)row0 * CC;
    unsigned lA = lds_off(&sA[0]);
    unsigned lB = lds_off(&sB[0]);
#pragma unroll
    for (int i = tid; i < 32 * CC / 8; i += 256) {
      async_copy16(lA + i * 16, srcA + i * 8);
      async_copy16(lB + i * 16, srcB + i * 8);
    }
    int b = row0 / PPB;                            // tiles never straddle batches
    sM[tid] = means[b * CC + tid];
    async_wait0();
  }
  __syncthreads();

  int wave = tid >> 5, lane = tid & 31;
  int msub  = wave & 1;                            // row sub-tile (0/1)
  int nbase = (wave >> 1) * 64;                    // 64 output cols per wave
  v8f acc[4] = {vzero8(), vzero8(), vzero8(), vzero8()};

  // ---- local geo features: K = 2048 over Wl ----
  for (int k0 = 0; k0 < 2048; k0 += 32) {
    for (int e = tid; e < 1024; e += 256) {
      int r = e >> 5, jj = e & 31;
      int j = k0 + jj;
      int si  = j >> 9;
      int typ = (j >> 8) & 1;
      int c   = j & 255;
      int cs  = (c - (1 << si)) & 255;
      float a  = bf2f(sA[r * CC + c]);
      float bs = bf2f(sB[r * CC + cs]);
      float as = bf2f(sA[r * CC + cs]);
      float bb = bf2f(sB[r * CC + c]);
      sF[r * 32 + jj] = f2bf(typ ? (a * bs - as * bb) : (a * bs + as * bb));
    }
    __syncthreads();
    v16bf a = load_a_frag(sF, 32, msub * 16, 0, lane);
#pragma unroll
    for (int t = 0; t < 4; ++t) {
      v16bf b = load_b_frag(wlT, 2048, nbase + 16 * t, k0, lane);
      acc[t] = __builtin_amdgcn_wmma_f32_16x16x32_bf16(false, a, false, b,
                                                       (short)0, acc[t], false, false);
    }
    __syncthreads();
  }

  // ---- global geo features: K = 1024 over Wg; b = c_glo = mean - zdet ----
  for (int k0 = 0; k0 < 1024; k0 += 32) {
    for (int e = tid; e < 1024; e += 256) {
      int r = e >> 5, jj = e & 31;
      int j = k0 + jj;
      int si  = j >> 9;                            // 0..1 -> shift 1,2
      int typ = (j >> 8) & 1;
      int c   = j & 255;
      int cs  = (c - (1 << si)) & 255;
      float a   = bf2f(sA[r * CC + c]);
      float as  = bf2f(sA[r * CC + cs]);
      float bcs = sM[cs] - as;                     // c_glo[cs]
      float bc  = sM[c] - a;                       // c_glo[c]
      sF[r * 32 + jj] = f2bf(typ ? (a * bcs - as * bc) : (a * bcs + as * bc));
    }
    __syncthreads();
    v16bf a = load_a_frag(sF, 32, msub * 16, 0, lane);
#pragma unroll
    for (int t = 0; t < 4; ++t) {
      v16bf b = load_b_frag(wgT, 1024, nbase + 16 * t, k0, lane);
      acc[t] = __builtin_amdgcn_wmma_f32_16x16x32_bf16(false, a, false, b,
                                                       (short)0, acc[t], false, false);
    }
    __syncthreads();
  }

  int khl = lane >> 4, nl = lane & 15;
#pragma unroll
  for (int t = 0; t < 4; ++t)
#pragma unroll
    for (int v = 0; v < 8; ++v) {
      int m = row0 + msub * 16 + v + 8 * khl;
      int n = nbase + 16 * t + nl;
      gfeat[(size_t)m * CC + n] = f2bf(acc[t][v]);
    }
}

// ---------------- kernel 7: gate GEMM + final residual combine ---------------
// alpha = sigmoid([xn | gfeat] @ w_gate); out = x + (silu(xn) + alpha*gfeat)*gamma_ls
// B tile (64 cols x 512 K, 64KB) staged to LDS once via async-to-LDS.

__global__ void __launch_bounds__(256)
k_gate(const float* __restrict__ x, const bf16* __restrict__ xn,
       const bf16* __restrict__ gfeat, const bf16* __restrict__ wgtT,
       const float* __restrict__ gls, float* __restrict__ out) {
  __shared__ bf16 sW[64 * 512];                     // 64KB
  int tid  = threadIdx.x;
  int wave = tid >> 5, lane = tid & 31;
  int row0 = blockIdx.x * 128 + wave * 16;
  int nc0  = blockIdx.y * 64;
  {
    const bf16* src = wgtT + (size_t)nc0 * 512;     // contiguous 64*512 bf16
    unsigned l0 = lds_off(&sW[0]);
#pragma unroll
    for (int i = tid; i < 64 * 512 / 8; i += 256)
      async_copy16(l0 + i * 16, src + i * 8);
    async_wait0();
  }
  __syncthreads();

  v8f acc[4] = {vzero8(), vzero8(), vzero8(), vzero8()};
  for (int k0 = 0; k0 < 512; k0 += 32) {
    const bf16* Asrc = (k0 < 256) ? xn : gfeat;    // concat along K
    int kk = k0 & 255;
    v16bf a = load_a_frag(Asrc, CC, row0, kk, lane);
#pragma unroll
    for (int t = 0; t < 4; ++t) {
      v16bf b = load_b_frag(sW, 512, 16 * t, k0, lane);  // from LDS
      acc[t] = __builtin_amdgcn_wmma_f32_16x16x32_bf16(false, a, false, b,
                                                       (short)0, acc[t], false, false);
    }
  }
  int khl = lane >> 4, nl = lane & 15;
#pragma unroll
  for (int t = 0; t < 4; ++t)
#pragma unroll
    for (int v = 0; v < 8; ++v) {
      int m = row0 + v + 8 * khl;
      int n = nc0 + 16 * t + nl;
      size_t idx = (size_t)m * CC + n;
      float alpha = 1.f / (1.f + __expf(-acc[t][v]));
      float xv = bf2f(xn[idx]);
      float gv = bf2f(gfeat[idx]);
      float si = xv / (1.f + __expf(-xv));         // silu(x_norm)
      out[idx] = x[idx] + (si + alpha * gv) * gls[n];
    }
}

// ---------------- host launcher ---------------------------------------------

extern "C" void kernel_launch(void* const* d_in, const int* in_sizes, int n_in,
                              void* d_out, int out_size, void* d_ws, size_t ws_size,
                              hipStream_t stream) {
  (void)in_sizes; (void)n_in; (void)out_size; (void)ws_size;
  const float* x      = (const float*)d_in[0];
  const float* ln_g   = (const float*)d_in[1];
  const float* w_det  = (const float*)d_in[2];
  const float* dw1    = (const float*)d_in[3];
  const float* dw2    = (const float*)d_in[4];
  const float* bn_g   = (const float*)d_in[5];
  const float* bn_m   = (const float*)d_in[6];
  const float* bn_v   = (const float*)d_in[7];
  const float* w_gl   = (const float*)d_in[8];
  const float* w_gg   = (const float*)d_in[9];
  const float* w_gate = (const float*)d_in[10];
  const float* g_ls   = (const float*)d_in[11];
  float* out = (float*)d_out;

  char* p = (char*)d_ws;
  auto take = [&](size_t bytes) -> char* {
    char* r = p;
    p += (bytes + 255) & ~(size_t)255;
    return r;
  };
  bf16*  xn    = (bf16*)take((size_t)NPIX * CC * 2);
  bf16*  zdet  = (bf16*)take((size_t)NPIX * CC * 2);
  bf16*  t1    = (bf16*)take((size_t)NPIX * CC * 2);   // conv tmp, reused as gfeat
  bf16*  zctx  = (bf16*)take((size_t)NPIX * CC * 2);
  bf16*  wdetT = (bf16*)take((size_t)CC * CC * 2);
  bf16*  wlT   = (bf16*)take((size_t)CC * 2048 * 2);
  bf16*  wgT   = (bf16*)take((size_t)CC * 1024 * 2);
  bf16*  wgtT  = (bf16*)take((size_t)CC * 512 * 2);
  float* means = (float*)take((size_t)NB * CC * 4);
  bf16*  gfeat = t1;

  k_prep<<<512, 256, 0, stream>>>(w_det, w_gl, w_gg, w_gate, wdetT, wlT, wgT, wgtT);
  k_ln<<<NPIX / 8, 256, 0, stream>>>(x, ln_g, xn);
  k_mean<<<NB * 16, 256, 0, stream>>>(xn, means);
  k_gemm_det<<<dim3(NPIX / 128, CC / 64), 256, 0, stream>>>(xn, wdetT, zdet);
  k_dw1<<<(NPIX * 64) / 256, 256, 0, stream>>>(xn, dw1, t1);
  k_dw2<<<(NPIX * 64) / 256, 256, 0, stream>>>(t1, dw2, bn_g, bn_m, bn_v, zdet, zctx);
  k_geo<<<NPIX / 32, 256, 0, stream>>>(zdet, zctx, means, wlT, wgT, gfeat);
  k_gate<<<dim3(NPIX / 128, CC / 64), 256, 0, stream>>>(x, xn, gfeat, wgtT, g_ls, out);
}
